// NaiveFourierKANLayer_60885456388481
// MI455X (gfx1250) — matrix operand order repaired and checked by reading, not compiled, not run on hardware
//
#include <hip/hip_runtime.h>

// ---------------------------------------------------------------------------
// Fourier-KAN layer as an f16-WMMA GEMM on gfx1250 (MI455X, wave32).
//   Y[8192,1024] = F[8192,16384] * Wt[16384,1024]
//   K index = cs*8192 + i*8 + g  (cs=0 -> cos features, cs=1 -> sin features)
//
// Kernel 1 (fkan_pack_w): W f32 [2,O,I,G] -> f16 B-fragment image in d_ws
//   (32 MB of workspace; resident in the 192 MB L2).
// Kernel 2 (fkan_gemm): 256 workgroups x 256 threads (8 waves).
//   Workgroup tile: 32 rows (b) x 1024 cols (j), K loop of 512 steps x 32.
//   Per step: all 8 B fragments are preloaded into registers (latency hides
//   under the transcendentals / barrier / previous WMMA chain), the x value
//   for the NEXT step is software-pipelined, features are computed BRANCH-FREE
//   (cos via sin(θ+π/2), one v_sin_f32 each) and land in LDS in the exact
//   per-lane A-fragment order k = 16*(lane/16) + elem, as in Bt (double-buffered), then
//   16 v_wmma_f32_16x16x32_f16 consume the fragments with staggered waits.
// ---------------------------------------------------------------------------

typedef __attribute__((ext_vector_type(16))) _Float16 v16h;
typedef __attribute__((ext_vector_type(8)))  _Float16 v8h;
typedef __attribute__((ext_vector_type(4)))  _Float16 v4h;
typedef __attribute__((ext_vector_type(8)))  float    v8f;

#define INPUTDIM 1024
#define OUTDIM   1024
#define BATCH    8192
#define GRIDSZ   8
#define KDIM     (2 * INPUTDIM * GRIDSZ)   // 16384
#define KSTEPS   (KDIM / 32)               // 512
#define NTILES   (OUTDIM / 16)             // 64
#define MBLK     32                        // rows per workgroup
#define MT       (MBLK / 16)               // 2 m-tiles per wave
#define NTW      8                         // n-tiles per wave (8 waves * 8 * 16 = 1024 cols)

#define HALF_PI 1.57079632679489662f

// --------------------------------------------------------------------------
// Pack W (f32 [2][O][I][G]) into B-fragment-ordered f16:
//   Bt[kstep][ntile][lane] = 32 bytes (16 f16) in this kernel's contiguous-k B order:
//     col n = ntile*16 + (lane&15)
//     VGPR v, slot s: k_in = 2v + s + (lane>=16 ? 16 : 0)
// Each thread produces one 16-byte half (8 contiguous K = one (cs,i), g=0..7),
// so reads are 32B-contiguous f32 and writes are 16B-contiguous f16.
// --------------------------------------------------------------------------
__global__ __launch_bounds__(256) void fkan_pack_w(const float* __restrict__ W,
                                                   v8h* __restrict__ Bt) {
    const int flat  = blockIdx.x * 256 + threadIdx.x;   // 2,097,152 threads
    const int h     =  flat        & 1;                 // which 16B half
    const int lane  = (flat >> 1)  & 31;
    const int ntile = (flat >> 6)  & 63;
    const int kstep =  flat >> 12;

    const int n    = ntile * 16 + (lane & 15);
    const int hi16 = (lane & 16) ? 16 : 0;
    const int K    = kstep * 32 + h * 8 + hi16;         // first of 8 contiguous K
    const int cs   = K >> 13;                           // 0 = cos half, 1 = sin half
    const int i    = (K & (8 * INPUTDIM - 1)) >> 3;     // input-dim index (g = 0..7)

    const float4* wp =
        (const float4*)(W + (size_t)((cs * OUTDIM + n) * INPUTDIM + i) * GRIDSZ);
    const float4 w0 = wp[0];
    const float4 w1 = wp[1];

    v8h o;
    o[0] = (_Float16)w0.x; o[1] = (_Float16)w0.y;
    o[2] = (_Float16)w0.z; o[3] = (_Float16)w0.w;
    o[4] = (_Float16)w1.x; o[5] = (_Float16)w1.y;
    o[6] = (_Float16)w1.z; o[7] = (_Float16)w1.w;
    Bt[flat] = o;
}

// --------------------------------------------------------------------------
// Fused feature-generation + WMMA GEMM.
// --------------------------------------------------------------------------
__global__ __launch_bounds__(256) void fkan_gemm(const float* __restrict__ x,
                                                 const v16h* __restrict__ Bt,
                                                 float* __restrict__ out) {
    // A-fragment staging: [buf][mtile][lane] -> 32B per lane, double-buffered.
    __shared__ v16h feat[2][MT][32];    // 4 KB

    const int tid    = threadIdx.x;
    const int lane   = tid & 31;
    const int wave   = tid >> 5;        // 0..7
    const int mblock = blockIdx.x * MBLK;

    // ---- feature-writer role: thread -> (m-tile, lane, quarter) -------------
    // Quarter wq covers VGPRs {2wq, 2wq+1} of the A fragment:
    //   k0 = 4*wq + (lane>=16 ? 16 : 0)
    // -> 4 contiguous k within one input index i (g-range 0..3 or 4..7).
    const int wt    =  tid >> 7;                       // m-tile 0..1
    const int wlane = (tid >> 2) & 31;
    const int wq    =  tid & 3;
    const int w_m   = mblock + wt * 16 + (wlane & 15); // batch row
    const int w_hi8 = (wlane & 16) ? 16 : 0;
    const int w_kq  = wq * 4;
    const int w_k0  = w_kq + w_hi8;                    // k within 32-chunk
    const int w_ioff = w_k0 >> 3;                      // 0..3
    const float g1   = (float)((w_k0 & 7) + 1);        // first harmonic (1 or 5)

    const float* xrow = x + (size_t)w_m * INPUTDIM;

    // ---- accumulators: 2 x 8 C-tiles of f32 (128 VGPRs) --------------------
    v8f acc[MT][NTW];
#pragma unroll
    for (int tm = 0; tm < MT; ++tm)
#pragma unroll
        for (int tn = 0; tn < NTW; ++tn)
#pragma unroll
            for (int v = 0; v < 8; ++v)
                acc[tm][tn][v] = 0.0f;

    // Software-pipelined x value (step 0 uses i = w_ioff).
    float xv = xrow[w_ioff];

    for (int step = 0; step < KSTEPS; ++step) {
        const int buf = step & 1;

        // ---- issue ALL B-fragment loads for this step first ----------------
        // (16x global_load_b128 in one clause; latency hides under the
        //  transcendentals + barrier below; Bt is L2-resident.)
        const v16h* bp = Bt + ((size_t)step * NTILES + wave * NTW) * 32 + lane;
        v16h bfrag[NTW];
#pragma unroll
        for (int tn = 0; tn < NTW; ++tn)
            bfrag[tn] = bp[tn * 32];

        // ---- software-pipelined x load for the NEXT step (issued last so
        //      waiting on it never drains the B loads above) -----------------
        const int snext = (step + 1 < KSTEPS) ? (step + 1) : 0;
        const float xv_next = xrow[((snext & 255) << 2) + w_ioff];

        // ---- branch-free features: cos(t) == sin(t + pi/2) -----------------
        // phase is uniform per step: pi/2 for the cos half (steps < 256),
        // 0 for the sin half -> exactly one v_sin_f32 per feature, no CF.
        const float phase = (step < (KSTEPS / 2)) ? HALF_PI : 0.0f;
        v4h f;
#pragma unroll
        for (int p = 0; p < 4; ++p) {
            const float th = xv * (g1 + (float)p) + phase;
            f[p] = (_Float16)__sinf(th);
        }
        ((v4h*)&feat[buf][wt][wlane])[wq] = f;           // ds_store_b64
        __syncthreads();                                  // features ready

        // ---- load A fragments (shared by all 8 waves) ----------------------
        v16h afrag[MT];
#pragma unroll
        for (int tm = 0; tm < MT; ++tm)
            afrag[tm] = feat[buf][tm][lane];             // 2x ds_load_b128 each

        // ---- 16 WMMAs; waits stagger as the load clause drains -------------
#pragma unroll
        for (int tn = 0; tn < NTW; ++tn) {
#pragma unroll
            for (int tm = 0; tm < MT; ++tm) {
                acc[tm][tn] = __builtin_amdgcn_wmma_f32_16x16x32_f16(
                    /*neg_a=*/false, afrag[tm],
                    /*neg_b=*/false, bfrag[tn],
                    /*c_mod=*/(short)0, acc[tm][tn],
                    /*reuse_a=*/false, /*reuse_b=*/false);
            }
        }

        xv = xv_next;
        // Double buffering: the single barrier above is sufficient — buf p is
        // rewritten two steps later, separated by two barriers.
    }

    // ---- write C tiles (f32 16x16 layout: lane<16 -> M=v, lane>=16 -> M=8+v)
#pragma unroll
    for (int tm = 0; tm < MT; ++tm) {
        const int row0 = mblock + tm * 16 + ((lane & 16) ? 8 : 0);
#pragma unroll
        for (int tn = 0; tn < NTW; ++tn) {
            const int col = wave * (NTW * 16) + tn * 16 + (lane & 15);
#pragma unroll
            for (int v = 0; v < 8; ++v)
                out[(size_t)(row0 + v) * OUTDIM + col] = acc[tm][tn][v];
        }
    }
}

// --------------------------------------------------------------------------
extern "C" void kernel_launch(void* const* d_in, const int* in_sizes, int n_in,
                              void* d_out, int out_size, void* d_ws, size_t ws_size,
                              hipStream_t stream) {
    (void)in_sizes; (void)n_in; (void)out_size; (void)ws_size;

    const float* x = (const float*)d_in[0];   // [8192,1024] f32
    const float* W = (const float*)d_in[1];   // [2,1024,1024,8] f32
    float* out     = (float*)d_out;           // [8192,1024] f32
    v8h* Bt        = (v8h*)d_ws;              // 32 MB packed f16 weights

    // 1) pack/convert weights: 512*64*32*2 = 2,097,152 threads
    fkan_pack_w<<<8192, 256, 0, stream>>>(W, Bt);

    // 2) fused feature-gen + WMMA GEMM: 8192/32 = 256 workgroups
    fkan_gemm<<<BATCH / MBLK, 256, 0, stream>>>(x, (const v16h*)Bt, out);
}
